// CausalMultiHeadAttention_2027224564559
// MI455X (gfx1250) — matrix operand order repaired and checked by reading, not compiled, not run on hardware
//
#include <hip/hip_runtime.h>

// ---------------- problem constants ----------------
#define DM   768
#define NH   12
#define DK   64
#define BB   4
#define SS   2048
#define SCALE 0.125f            // 1/sqrt(64)
#define MSEQ (BB * SS)          // 8192 rows for projection GEMMs

// ---------------- CDNA5 WMMA types -----------------
typedef __bf16 bf16;
typedef __attribute__((ext_vector_type(16))) __bf16 v16bf;
typedef __attribute__((ext_vector_type(8)))  __bf16 v8bf;
typedef __attribute__((ext_vector_type(4)))  __bf16 v4bf;
typedef __attribute__((ext_vector_type(8)))  float  v8f;

union AB16 { v16bf v; v8bf h[2]; };

__device__ __forceinline__ v8f wmma_bf16(v16bf a, v16bf b, v8f c) {
    return __builtin_amdgcn_wmma_f32_16x16x32_bf16(
        false, a, false, b, (short)0, c, false, false);
}

// ---- gfx1250 async global->LDS copy (16B per lane, ASYNCcnt tracked) ----
__device__ __forceinline__ void async_ld16(void* lds, const void* gsrc) {
    unsigned l = (unsigned)(uintptr_t)lds;   // low 32b of generic LDS ptr == LDS offset
    asm volatile("global_load_async_to_lds_b128 %0, %1, off"
                 :: "v"(l), "v"(gsrc) : "memory");
}
__device__ __forceinline__ void async_wait0() {
    asm volatile("s_wait_asynccnt 0x0" ::: "memory");
}

// =====================================================================
// one-time converters
// =====================================================================
__global__ __launch_bounds__(256)
void mha_cvt_bf16(const float* __restrict__ s, bf16* __restrict__ d, int n4)
{
    int i = blockIdx.x * 256 + threadIdx.x;
    if (i < n4) {
        float4 f = ((const float4*)s)[i];
        v4bf o;
        o[0] = (bf16)f.x; o[1] = (bf16)f.y; o[2] = (bf16)f.z; o[3] = (bf16)f.w;
        *(v4bf*)(d + (size_t)i * 4) = o;
    }
}

__global__ __launch_bounds__(256)
void mha_wt_bf16(const float* __restrict__ W, bf16* __restrict__ Wt)
{
    // W (K,N) fp32 -> Wt (N,K) bf16, K = N = DM
    int idx = blockIdx.x * 256 + threadIdx.x;     // DM*DM total
    int n = idx / DM, k = idx - n * DM;
    Wt[(size_t)n * DM + k] = (bf16)W[(size_t)k * DM + n];
}

// =====================================================================
// GEMM: Y[M,N] = Xb[M,K](bf16) @ Wt(N,K)(bf16) + bias, ping-pong async LDS
// block 256 = 8 waves; workgroup tile 128x64; wave tile 32x32 (2x2 WMMA)
// =====================================================================
#define TM 128
#define TN 64
#define TK 32

__global__ __launch_bounds__(256)
void mha_gemm_bias(const bf16* __restrict__ X, const bf16* __restrict__ Wt,
                   const float* __restrict__ bias,
                   float* __restrict__ Yf, bf16* __restrict__ Yb,
                   int M, int N, int K)
{
    __shared__ bf16 As[2][TM][TK];   // 2 x 8 KB
    __shared__ bf16 Bs[2][TN][TK];   // 2 x 4 KB

    const int tid  = threadIdx.x;
    const int wid  = tid >> 5;
    const int lane = tid & 31;
    const int l16  = lane & 15;
    const int lh   = lane >> 4;
    const int m0   = blockIdx.x * TM;
    const int n0   = blockIdx.y * TN;
    const int wm   = (wid & 3) * 32;
    const int wn   = (wid >> 2) * 32;

    // staging coordinates (fixed per thread)
    const int ar0 = (tid * 8) >> 5,        ac0 = (tid * 8) & 31;
    const int ar1 = ((tid + 256) * 8) >> 5, ac1 = ((tid + 256) * 8) & 31;
    const int bn0 = (tid * 8) >> 5,        bc0 = (tid * 8) & 31;

    auto stage = [&](int buf, int k0) {
        async_ld16(&As[buf][ar0][ac0], &X[(size_t)(m0 + ar0) * K + (k0 + ac0)]);
        async_ld16(&As[buf][ar1][ac1], &X[(size_t)(m0 + ar1) * K + (k0 + ac1)]);
        async_ld16(&Bs[buf][bn0][bc0], &Wt[(size_t)(n0 + bn0) * K + (k0 + bc0)]);
    };

    v8f acc[2][2] = {};
    const int NKT = K / TK;
    stage(0, 0);

    for (int i = 0; i < NKT; ++i) {
        const int buf = i & 1;
        async_wait0();          // tile i resident in LDS
        __syncthreads();        // all waves see it; prev readers of buf^1 done
        if (i + 1 < NKT) stage(buf ^ 1, (i + 1) * TK);

        #pragma unroll
        for (int mi = 0; mi < 2; ++mi) {
            const bf16* ar = &As[buf][wm + mi * 16 + l16][0];
            AB16 af;
            af.h[0] = *(const v8bf*)(ar + lh * 16);
            af.h[1] = *(const v8bf*)(ar + lh * 16 + 8);
            #pragma unroll
            for (int ni = 0; ni < 2; ++ni) {
                v16bf bf = *(const v16bf*)&Bs[buf][wn + ni * 16 + l16][lh * 16];
                acc[mi][ni] = wmma_bf16(af.v, bf, acc[mi][ni]);
            }
        }
        __syncthreads();        // readers done before buf is restaged
    }

    // epilogue: uniform branch hoisted out of the store loops
    if (Yf != nullptr) {
        #pragma unroll
        for (int mi = 0; mi < 2; ++mi)
            #pragma unroll
            for (int ni = 0; ni < 2; ++ni) {
                int col = n0 + wn + ni * 16 + l16;
                float b = bias[col];
                #pragma unroll
                for (int r = 0; r < 8; ++r) {
                    int row = m0 + wm + mi * 16 + lh * 8 + r;
                    Yf[(size_t)row * N + col] = acc[mi][ni][r] + b;
                }
            }
    } else {
        #pragma unroll
        for (int mi = 0; mi < 2; ++mi)
            #pragma unroll
            for (int ni = 0; ni < 2; ++ni) {
                int col = n0 + wn + ni * 16 + l16;
                float b = bias[col];
                #pragma unroll
                for (int r = 0; r < 8; ++r) {
                    int row = m0 + wm + mi * 16 + lh * 8 + r;
                    Yb[(size_t)row * N + col] = (bf16)(acc[mi][ni][r] + b);
                }
            }
    }
}

// =====================================================================
// Attention pass 1: per-row causal softmax stats (max m, denom l)
// grid = (S/128, H, B), block 256 (wave owns 16 query rows)
// =====================================================================
__global__ __launch_bounds__(256)
void mha_attn_stats(const bf16* __restrict__ Q, const bf16* __restrict__ Km,
                    float* __restrict__ mbuf, float* __restrict__ lbuf)
{
    __shared__ bf16 Qs[128][DK];     // 16 KB
    __shared__ bf16 Ks[2][32][DK];   // 2 x 4 KB

    const int tid  = threadIdx.x;
    const int wid  = tid >> 5;
    const int lane = tid & 31;
    const int l16  = lane & 15;
    const int lh   = lane >> 4;
    const int q0   = blockIdx.x * 128;
    const int h    = blockIdx.y;
    const int b    = blockIdx.z;
    const int qend = q0 + 128;

    const int kr = (tid * 8) >> 6, kc = (tid * 8) & 63;
    auto stageK = [&](int buf, int kt) {
        async_ld16(&Ks[buf][kr][kc],
                   &Km[(size_t)(b * SS + kt + kr) * DM + h * DK + kc]);
    };

    // stage Q block: 16 KB -> 4 async chunks per thread
    #pragma unroll
    for (int it = 0; it < 4; ++it) {
        int e = (tid + it * 256) * 8;
        int r = e >> 6, c = e & 63;
        async_ld16(&Qs[r][c], &Q[(size_t)(b * SS + q0 + r) * DM + h * DK + c]);
    }
    stageK(0, 0);
    async_wait0();
    __syncthreads();

    AB16 aq[2];
    {
        const bf16* qr = &Qs[wid * 16 + l16][0];
        #pragma unroll
        for (int ks = 0; ks < 2; ++ks) {
            aq[ks].h[0] = *(const v8bf*)(qr + ks * 32 + lh * 16);
            aq[ks].h[1] = *(const v8bf*)(qr + ks * 32 + lh * 16 + 8);
        }
    }

    float m[8], l[8];
    #pragma unroll
    for (int r = 0; r < 8; ++r) { m[r] = -1e30f; l[r] = 0.f; }

    const int NKT = qend / 32;
    for (int i = 0; i < NKT; ++i) {
        const int buf = i & 1;
        if (i) { async_wait0(); __syncthreads(); }
        if (i + 1 < NKT) stageK(buf ^ 1, (i + 1) * 32);

        #pragma unroll
        for (int sub = 0; sub < 2; ++sub) {
            v8f c = {};
            #pragma unroll
            for (int ks = 0; ks < 2; ++ks) {
                v16bf bf = *(const v16bf*)&Ks[buf][sub * 16 + l16][ks * 32 + lh * 16];
                c = wmma_bf16(aq[ks].v, bf, c);
            }
            int colg = i * 32 + sub * 16 + l16;
            #pragma unroll
            for (int r = 0; r < 8; ++r) {
                int rowg = q0 + wid * 16 + lh * 8 + r;
                float val = (colg <= rowg) ? c[r] * SCALE : -1e30f;
                float tmax = val;
                #pragma unroll
                for (int off = 8; off >= 1; off >>= 1)
                    tmax = fmaxf(tmax, __shfl_xor(tmax, off, 32));
                float nm = fmaxf(m[r], tmax);
                float p  = __expf(val - nm);
                float ts = p;
                #pragma unroll
                for (int off = 8; off >= 1; off >>= 1)
                    ts += __shfl_xor(ts, off, 32);
                l[r] = l[r] * __expf(m[r] - nm) + ts;
                m[r] = nm;
            }
        }
        __syncthreads();    // retire readers of buf before it is restaged
    }

    if (l16 == 0) {
        size_t base = ((size_t)(b * NH + h)) * SS + q0 + wid * 16 + lh * 8;
        #pragma unroll
        for (int r = 0; r < 8; ++r) { mbuf[base + r] = m[r]; lbuf[base + r] = l[r]; }
    }
}

// =====================================================================
// Attention pass 2: normalized P -> attention_weights (fp32) + attn = P@V
// =====================================================================
__global__ __launch_bounds__(256)
void mha_attn_pv(const bf16* __restrict__ Q, const bf16* __restrict__ Km,
                 const bf16* __restrict__ V,
                 const float* __restrict__ mbuf, const float* __restrict__ lbuf,
                 float* __restrict__ AW, bf16* __restrict__ attnb)
{
    __shared__ bf16 Qs[128][DK];       // 16 KB
    __shared__ bf16 Ks[2][32][DK];     //  8 KB (key, d)
    __shared__ bf16 Vs[2][DK][32];     //  8 KB (d, key) transposed
    __shared__ bf16 Ps[8][16][32];     //  8 KB per-wave P tile

    const int tid  = threadIdx.x;
    const int wid  = tid >> 5;
    const int lane = tid & 31;
    const int l16  = lane & 15;
    const int lh   = lane >> 4;
    const int q0   = blockIdx.x * 128;
    const int h    = blockIdx.y;
    const int b    = blockIdx.z;
    const int qend = q0 + 128;

    const int kr = (tid * 8) >> 6, kc = (tid * 8) & 63;
    auto stageKV = [&](int buf, int kt) {
        size_t g = (size_t)(b * SS + kt + kr) * DM + h * DK + kc;
        async_ld16(&Ks[buf][kr][kc], &Km[g]);
        v8bf vv = *(const v8bf*)&V[g];          // V tile, transposed LDS store
        #pragma unroll
        for (int j = 0; j < 8; ++j)
            Vs[buf][kc + j][kr] = vv[j];
    };

    #pragma unroll
    for (int it = 0; it < 4; ++it) {
        int e = (tid + it * 256) * 8;
        int r = e >> 6, c = e & 63;
        async_ld16(&Qs[r][c], &Q[(size_t)(b * SS + q0 + r) * DM + h * DK + c]);
    }
    stageKV(0, 0);
    async_wait0();
    __syncthreads();

    AB16 aq[2];
    {
        const bf16* qr = &Qs[wid * 16 + l16][0];
        #pragma unroll
        for (int ks = 0; ks < 2; ++ks) {
            aq[ks].h[0] = *(const v8bf*)(qr + ks * 32 + lh * 16);
            aq[ks].h[1] = *(const v8bf*)(qr + ks * 32 + lh * 16 + 8);
        }
    }

    float mrow[8], rli[8];
    {
        size_t base = ((size_t)(b * NH + h)) * SS + q0 + wid * 16 + lh * 8;
        #pragma unroll
        for (int r = 0; r < 8; ++r) {
            mrow[r] = mbuf[base + r];
            rli[r]  = 1.0f / lbuf[base + r];
        }
    }

    v8f acc[4] = {};

    const int NKT = qend / 32;
    for (int i = 0; i < NKT; ++i) {
        const int buf = i & 1;
        if (i) { async_wait0(); __syncthreads(); }
        if (i + 1 < NKT) stageKV(buf ^ 1, (i + 1) * 32);

        // scores + normalized P: write AW fp32 + Ps bf16
        #pragma unroll
        for (int sub = 0; sub < 2; ++sub) {
            v8f c = {};
            #pragma unroll
            for (int ks = 0; ks < 2; ++ks) {
                v16bf bf = *(const v16bf*)&Ks[buf][sub * 16 + l16][ks * 32 + lh * 16];
                c = wmma_bf16(aq[ks].v, bf, c);
            }
            int colg = i * 32 + sub * 16 + l16;
            #pragma unroll
            for (int r = 0; r < 8; ++r) {
                int rowg = q0 + wid * 16 + lh * 8 + r;
                float p = 0.f;
                if (colg <= rowg)
                    p = __expf(c[r] * SCALE - mrow[r]) * rli[r];
                AW[(((size_t)(b * NH + h) * SS + rowg) * SS) + colg] = p;
                Ps[wid][lh * 8 + r][sub * 16 + l16] = (bf16)p;
            }
        }
        __syncthreads();   // Ps visible (and Ks/Vs readers retired)

        // attn += P(16x32) @ V(32x64)
        AB16 ap;
        ap.h[0] = *(const v8bf*)&Ps[wid][l16][lh * 16];
        ap.h[1] = *(const v8bf*)&Ps[wid][l16][lh * 16 + 8];
        #pragma unroll
        for (int ni = 0; ni < 4; ++ni) {
            v16bf bv = *(const v16bf*)&Vs[buf][ni * 16 + l16][lh * 16];
            acc[ni] = wmma_bf16(ap.v, bv, acc[ni]);
        }
        __syncthreads();   // retire Vs/Ps readers before restage/overwrite
    }

    // store attn as bf16 (input of the output projection)
    #pragma unroll
    for (int ni = 0; ni < 4; ++ni)
        #pragma unroll
        for (int r = 0; r < 8; ++r) {
            int rowg = q0 + wid * 16 + lh * 8 + r;
            attnb[(size_t)(b * SS + rowg) * DM + h * DK + ni * 16 + l16] =
                (bf16)acc[ni][r];
        }

    // zero the never-touched causal tail columns [qend, S)
    for (int row = 0; row < 128; ++row) {
        size_t base = ((size_t)(b * NH + h) * SS + (q0 + row)) * SS;
        for (int c = qend + tid; c < SS; c += 256)
            AW[base + c] = 0.f;
    }
}

// =====================================================================
// host launcher
// =====================================================================
extern "C" void kernel_launch(void* const* d_in, const int* in_sizes, int n_in,
                              void* d_out, int out_size, void* d_ws, size_t ws_size,
                              hipStream_t stream)
{
    const float* x  = (const float*)d_in[0];
    const float* Wq = (const float*)d_in[1];
    const float* bq = (const float*)d_in[2];
    const float* Wk = (const float*)d_in[3];
    const float* bk = (const float*)d_in[4];
    const float* Wv = (const float*)d_in[5];
    const float* bv = (const float*)d_in[6];
    const float* Wo = (const float*)d_in[7];
    const float* bo = (const float*)d_in[8];

    float* out = (float*)d_out;                   // (B,S,DM) fp32
    float* AW  = out + (size_t)BB * SS * DM;      // (B,H,S,S) fp32

    const size_t NX = (size_t)BB * SS * DM;       // 6,291,456 elements
    const size_t NW = (size_t)DM * DM;            // 589,824 elements
    bf16* xb   = (bf16*)d_ws;
    bf16* Qb   = xb  + NX;
    bf16* Kb   = Qb  + NX;
    bf16* Vb   = Kb  + NX;
    bf16* Ab   = Vb  + NX;
    bf16* Wtq  = Ab  + NX;
    bf16* Wtk  = Wtq + NW;
    bf16* Wtv  = Wtk + NW;
    bf16* Wto  = Wtv + NW;
    float* mbuf = (float*)(Wto + NW);
    float* lbuf = mbuf + (size_t)BB * NH * SS;

    // one-time conversions
    mha_cvt_bf16<<<(int)(NX / 4 / 256), 256, 0, stream>>>(x, xb, (int)(NX / 4));
    const int gW = (int)(NW / 256);
    mha_wt_bf16<<<gW, 256, 0, stream>>>(Wq, Wtq);
    mha_wt_bf16<<<gW, 256, 0, stream>>>(Wk, Wtk);
    mha_wt_bf16<<<gW, 256, 0, stream>>>(Wv, Wtv);
    mha_wt_bf16<<<gW, 256, 0, stream>>>(Wo, Wto);

    // projections -> bf16 Q/K/V
    dim3 gGemm(MSEQ / TM, DM / TN);               // 64 x 12
    mha_gemm_bias<<<gGemm, 256, 0, stream>>>(xb, Wtq, bq, nullptr, Qb, MSEQ, DM, DM);
    mha_gemm_bias<<<gGemm, 256, 0, stream>>>(xb, Wtk, bk, nullptr, Kb, MSEQ, DM, DM);
    mha_gemm_bias<<<gGemm, 256, 0, stream>>>(xb, Wtv, bv, nullptr, Vb, MSEQ, DM, DM);

    dim3 gAttn(SS / 128, NH, BB);                 // 16 x 12 x 4
    mha_attn_stats<<<gAttn, 256, 0, stream>>>(Qb, Kb, mbuf, lbuf);
    mha_attn_pv<<<gAttn, 256, 0, stream>>>(Qb, Kb, Vb, mbuf, lbuf, AW, Ab);

    // output projection -> fp32
    mha_gemm_bias<<<gGemm, 256, 0, stream>>>(Ab, Wto, bo, out, nullptr, MSEQ, DM, DM);
}
